// Block_25950192402667
// MI455X (gfx1250) — matrix operand order repaired and checked by reading, not compiled, not run on hardware
//
#include <hip/hip_runtime.h>
#include <hip/hip_bf16.h>
#include <math.h>

// ---------------- problem constants ----------------
#define Bb 4
#define Tt 1024
#define Cc 1024
#define Hh 16
#define Dd 64
#define NT (Bb * Tt)        // 4096 token rows
#define C3 (3 * Cc)         // 3072
#define C4 (4 * Cc)         // 4096

typedef __attribute__((ext_vector_type(16))) _Float16 v16h;
typedef __attribute__((ext_vector_type(8)))  _Float16 v8h;
typedef __attribute__((ext_vector_type(8)))  float    v8f;

// ---------- WMMA fragment helpers (wave32, 16x16x32 f16) ----------
// A (16xK=32): lane m = lane&15; VGPR v=g*4+r holds K = g*16 + half*8 + 2r
__device__ __forceinline__ v16h ld_a(const _Float16* __restrict__ p, int hf) {
  v8h lo = *(const v8h*)(p + hf * 8);
  v8h hi = *(const v8h*)(p + 16 + hf * 8);
  v16h r;
#pragma unroll
  for (int i = 0; i < 8; ++i) { r[i] = lo[i]; r[8 + i] = hi[i]; }
  return r;
}
// B (K=32 x 16): lane n = lane&15; VGPR v=g*4+r holds K = g*16 + half*8 + 2r (same as A)
__device__ __forceinline__ v16h ld_b(const _Float16* __restrict__ p, int hf) {
  v8h lo = *(const v8h*)(p + hf * 8);
  v8h hi = *(const v8h*)(p + 16 + hf * 8);
  v16h r;
#pragma unroll
  for (int i = 0; i < 8; ++i) { r[i] = lo[i]; r[8 + i] = hi[i]; }
  return r;
}
__device__ __forceinline__ v8f wmma16(v16h a, v16h b, v8f c) {
  return __builtin_amdgcn_wmma_f32_16x16x32_f16(false, a, false, b, (short)0, c,
                                                false, false);
}

__device__ __forceinline__ float gelu_exact(float v) {
  return 0.5f * v * (1.0f + erff(v * 0.70710678118654752f));
}

// ---------------- fp32 -> fp16 conversion ----------------
__global__ __launch_bounds__(256) void cvt_kernel(const float* __restrict__ src,
                                                  _Float16* __restrict__ dst, int n) {
  int i = blockIdx.x * 256 + threadIdx.x;
  int stride = gridDim.x * 256;
  for (; i < n; i += stride) dst[i] = (_Float16)src[i];
}

// ---------------- LayerNorm (bias-free), fp32 in -> fp16 out ----------------
__global__ __launch_bounds__(256) void ln_kernel(const float* __restrict__ x,
                                                 const float* __restrict__ w,
                                                 _Float16* __restrict__ out) {
  __shared__ float red[256];
  const int row = blockIdx.x;
  const int tid = threadIdx.x;
  const float* xr = x + (size_t)row * Cc;
  float v[4];
  float s = 0.f;
#pragma unroll
  for (int i = 0; i < 4; ++i) { v[i] = xr[tid + i * 256]; s += v[i]; }
  red[tid] = s;
  __syncthreads();
  for (int off = 128; off > 0; off >>= 1) {
    if (tid < off) red[tid] += red[tid + off];
    __syncthreads();
  }
  const float mu = red[0] * (1.f / Cc);
  __syncthreads();
  float s2 = 0.f;
#pragma unroll
  for (int i = 0; i < 4; ++i) { float d = v[i] - mu; s2 += d * d; }
  red[tid] = s2;
  __syncthreads();
  for (int off = 128; off > 0; off >>= 1) {
    if (tid < off) red[tid] += red[tid + off];
    __syncthreads();
  }
  const float rs = rsqrtf(red[0] * (1.f / Cc) + 1e-5f);
  _Float16* orow = out + (size_t)row * Cc;
#pragma unroll
  for (int i = 0; i < 4; ++i) {
    int c = tid + i * 256;
    orow[c] = (_Float16)((v[i] - mu) * rs * w[c]);
  }
}

// ---------------- tiled WMMA GEMM:  Y[M,N] = A[M,K] * W[N,K]^T ----------------
// mode 0: outF = acc + residual (f32)
// mode 1: outH = acc (f16)
// mode 2: outH = gelu(acc) (f16)
__global__ __launch_bounds__(256) void gemm_kernel(
    const _Float16* __restrict__ A, const _Float16* __restrict__ W,
    const float* __restrict__ residual, float* __restrict__ outF,
    _Float16* __restrict__ outH, int Kdim, int N, int mode) {
  const int tid = threadIdx.x, lane = tid & 31, wv = tid >> 5;
  const int wm = wv & 1, wn = wv >> 1;                 // 2 waves in M, 4 in N
  const int mrow = blockIdx.y * 128 + wm * 64;
  const int ncol = blockIdx.x * 128 + wn * 32;
  const int ln = lane & 15, hf = lane >> 4;

  v8f acc[4][2] = {};
  const _Float16* Arow[4];
  const _Float16* Wrow[2];
#pragma unroll
  for (int i = 0; i < 4; ++i) Arow[i] = A + (size_t)(mrow + i * 16 + ln) * Kdim;
#pragma unroll
  for (int j = 0; j < 2; ++j) Wrow[j] = W + (size_t)(ncol + j * 16 + ln) * Kdim;

  for (int k0 = 0; k0 < Kdim; k0 += 32) {
    if (k0 + 128 < Kdim) {
      __builtin_prefetch(Arow[0] + k0 + 128, 0, 1);   // global_prefetch_b8
      __builtin_prefetch(Wrow[0] + k0 + 128, 0, 1);
    }
    v16h a[4], b[2];
#pragma unroll
    for (int i = 0; i < 4; ++i) a[i] = ld_a(Arow[i] + k0, hf);
#pragma unroll
    for (int j = 0; j < 2; ++j) b[j] = ld_b(Wrow[j] + k0, hf);
#pragma unroll
    for (int i = 0; i < 4; ++i)
#pragma unroll
      for (int j = 0; j < 2; ++j) acc[i][j] = wmma16(a[i], b[j], acc[i][j]);
  }

#pragma unroll
  for (int i = 0; i < 4; ++i)
#pragma unroll
    for (int j = 0; j < 2; ++j)
#pragma unroll
      for (int v = 0; v < 8; ++v) {
        const int gr = mrow + i * 16 + v + hf * 8;
        const int gc = ncol + j * 16 + ln;
        const size_t idx = (size_t)gr * N + gc;
        const float val = acc[i][j][v];
        if (mode == 0) {
          outF[idx] = val + (residual ? residual[idx] : 0.f);
        } else if (mode == 1) {
          outH[idx] = (_Float16)val;
        } else {
          outH[idx] = (_Float16)gelu_exact(val);
        }
      }
}

// ---------------- top-2 sparse causal attention (fused, two-pass) ----------------
// qkv: [NT, 3C] fp16 (q | k | v, head h at cols h*64..h*64+63 of each third)
// y:   [NT, C]  fp16
__global__ __launch_bounds__(256) void attn_topk_kernel(
    const _Float16* __restrict__ qkv, _Float16* __restrict__ y) {
  const int bh = blockIdx.y;
  const int b = bh / Hh, h = bh % Hh;
  const int tid = threadIdx.x, lane = tid & 31, wv = tid >> 5;
  const int ln = lane & 15, hf = lane >> 4;
  const int row0 = blockIdx.x * 128 + wv * 16;   // this wave's 16 query rows

  const _Float16* qbase = qkv + (size_t)b * Tt * C3 + h * Dd;
  const _Float16* kbase = qbase + Cc;
  const _Float16* vbase = qbase + 2 * Cc;

  __shared__ __attribute__((aligned(16))) _Float16 Vt[Dd][72];       // V^T tile
  __shared__ __attribute__((aligned(16))) _Float16 Pb[8][16][40];    // per-wave P

  // Q fragments (D=64 -> two K=32 chunks), kept in registers
  v16h qa[2];
#pragma unroll
  for (int c = 0; c < 2; ++c)
    qa[c] = ld_a(qbase + (size_t)(row0 + ln) * C3 + c * 32, hf);

  // ---- pass 1: per-row top-2 over the FULL row (pre-causal) ----
  float t1[8], t2[8];
#pragma unroll
  for (int v = 0; v < 8; ++v) { t1[v] = -3.0e38f; t2[v] = -3.0e38f; }

  for (int kk = 0; kk < Tt; kk += 16) {
    const _Float16* kp = kbase + (size_t)(kk + ln) * C3;
    v8f s = {};
    s = wmma16(qa[0], ld_b(kp, hf), s);
    s = wmma16(qa[1], ld_b(kp + 32, hf), s);
#pragma unroll
    for (int v = 0; v < 8; ++v) {
      float val = s[v] * 0.125f;                      // 1/sqrt(64)
      if (val > t1[v]) { t2[v] = t1[v]; t1[v] = val; }
      else if (val > t2[v]) t2[v] = val;
    }
  }
  // merge top-2 across the 16 lanes of each half (rows disjoint per half)
#pragma unroll
  for (int off = 1; off < 16; off <<= 1) {
#pragma unroll
    for (int v = 0; v < 8; ++v) {
      float o1 = __shfl_xor(t1[v], off, 32);
      float o2 = __shfl_xor(t2[v], off, 32);
      float n1 = fmaxf(t1[v], o1);
      float n2 = fmaxf(fminf(t1[v], o1), fmaxf(t2[v], o2));
      t1[v] = n1; t2[v] = n2;
    }
  }

  // ---- pass 2: threshold + causal mask + softmax + P·V ----
  v8f yacc[4] = {};
  float dsum[8] = {};

  for (int kc = 0; kc < Tt; kc += 64) {
    __syncthreads();
    // stage V[kc..kc+64) transposed into LDS
    for (int i = tid; i < 64 * 64; i += 256) {
      int key = i >> 6, d = i & 63;
      Vt[d][key] = vbase[(size_t)(kc + key) * C3 + d];
    }
    __syncthreads();

#pragma unroll
    for (int sub = 0; sub < 2; ++sub) {             // two 32-key chunks
      const int k0 = kc + sub * 32;
#pragma unroll
      for (int half16 = 0; half16 < 2; ++half16) {  // two 16-key score tiles
        const int kk = k0 + half16 * 16;
        const _Float16* kp = kbase + (size_t)(kk + ln) * C3;
        v8f s = {};
        s = wmma16(qa[0], ld_b(kp, hf), s);
        s = wmma16(qa[1], ld_b(kp + 32, hf), s);
        const int key = kk + ln;
#pragma unroll
        for (int v = 0; v < 8; ++v) {
          float val = s[v] * 0.125f;
          int qrow = row0 + v + hf * 8;
          float p = (val >= t2[v] && key <= qrow) ? __expf(val - t1[v]) : 0.f;
          dsum[v] += p;
          Pb[wv][v + hf * 8][half16 * 16 + ln] = (_Float16)p;
        }
      }
      // reload P (D-layout -> A-layout via LDS; in-order DS within wave)
      v16h pa = ld_a(&Pb[wv][ln][0], hf);
#pragma unroll
      for (int j = 0; j < 4; ++j) {
        v16h vb = ld_b(&Vt[j * 16 + ln][sub * 32], hf);
        yacc[j] = wmma16(pa, vb, yacc[j]);
      }
    }
  }

  // reduce denominators across the 16 lanes of each half
#pragma unroll
  for (int off = 1; off < 16; off <<= 1)
#pragma unroll
    for (int v = 0; v < 8; ++v) dsum[v] += __shfl_xor(dsum[v], off, 32);

#pragma unroll
  for (int j = 0; j < 4; ++j)
#pragma unroll
    for (int v = 0; v < 8; ++v) {
      const int qrow = row0 + v + hf * 8;
      const float dn = dsum[v];
      const float val = (dn > 0.f) ? (yacc[j][v] / dn) : 0.f;
      y[(size_t)(b * Tt + qrow) * Cc + h * Dd + j * 16 + ln] = (_Float16)val;
    }
}

// ---------------- host orchestration ----------------
extern "C" void kernel_launch(void* const* d_in, const int* in_sizes, int n_in,
                              void* d_out, int out_size, void* d_ws, size_t ws_size,
                              hipStream_t stream) {
  (void)in_sizes; (void)n_in; (void)out_size; (void)ws_size;
  const float* x    = (const float*)d_in[0];
  const float* ln1w = (const float*)d_in[1];
  const float* wat  = (const float*)d_in[2];   // [3C, C]
  const float* watp = (const float*)d_in[3];   // [C, C]
  const float* ln2w = (const float*)d_in[4];
  const float* wfc  = (const float*)d_in[5];   // [4C, C]
  const float* wpr  = (const float*)d_in[6];   // [C, 4C]
  float* out = (float*)d_out;

  char* base = (char*)d_ws;
  size_t off = 0;
  auto alloc = [&](size_t bytes) -> void* {
    void* p = base + off;
    off += bytes;
    off = (off + 255) & ~(size_t)255;
    return p;
  };
  _Float16* h1   = (_Float16*)alloc((size_t)NT * Cc * 2);
  _Float16* waH  = (_Float16*)alloc((size_t)C3 * Cc * 2);
  _Float16* wpH  = (_Float16*)alloc((size_t)Cc * Cc * 2);
  _Float16* wfH  = (_Float16*)alloc((size_t)C4 * Cc * 2);
  _Float16* wp2H = (_Float16*)alloc((size_t)Cc * C4 * 2);
  _Float16* qkvH = (_Float16*)alloc((size_t)NT * C3 * 2);
  _Float16* yH   = (_Float16*)alloc((size_t)NT * Cc * 2);
  float*    x1   = (float*)   alloc((size_t)NT * Cc * 4);
  _Float16* h2   = (_Float16*)alloc((size_t)NT * Cc * 2);
  _Float16* fcH  = (_Float16*)alloc((size_t)NT * C4 * 2);

  // weight conversions fp32 -> fp16
  cvt_kernel<<<2048, 256, 0, stream>>>(wat,  waH,  C3 * Cc);
  cvt_kernel<<<2048, 256, 0, stream>>>(watp, wpH,  Cc * Cc);
  cvt_kernel<<<2048, 256, 0, stream>>>(wfc,  wfH,  C4 * Cc);
  cvt_kernel<<<2048, 256, 0, stream>>>(wpr,  wp2H, Cc * C4);

  // x1 = x + attn_proj(attn(ln1(x)))
  ln_kernel<<<NT, 256, 0, stream>>>(x, ln1w, h1);
  gemm_kernel<<<dim3(C3 / 128, NT / 128), 256, 0, stream>>>(
      h1, waH, nullptr, nullptr, qkvH, Cc, C3, 1);
  attn_topk_kernel<<<dim3(Tt / 128, Bb * Hh), 256, 0, stream>>>(qkvH, yH);
  gemm_kernel<<<dim3(Cc / 128, NT / 128), 256, 0, stream>>>(
      yH, wpH, x, x1, nullptr, Cc, Cc, 0);

  // out = x1 + proj(gelu(fc(ln2(x1))))
  ln_kernel<<<NT, 256, 0, stream>>>(x1, ln2w, h2);
  gemm_kernel<<<dim3(C4 / 128, NT / 128), 256, 0, stream>>>(
      h2, wfH, nullptr, nullptr, fcH, Cc, C4, 2);
  gemm_kernel<<<dim3(Cc / 128, NT / 128), 256, 0, stream>>>(
      fcH, wp2H, x1, out, nullptr, C4, Cc, 0);
}
